// RadialSymmetricConv3D_78365973283264
// MI455X (gfx1250) — matrix operand order repaired and checked by reading, not compiled, not run on hardware
//
#include <hip/hip_runtime.h>

// ---------------------------------------------------------------------------
// RadialSymmetricConv3D on MI455X (gfx1250, wave32, WMMA).
//
// out = conv3d(reflect_pad(x,12), PSF), PSF built from PSF_half[25,36] via
// radial gather+lerp, normalized by its global sum.
//
// Compute-bound (262 GFLOP vs 67 MB moved) -> bf16 WMMA implicit GEMM:
// the x-direction conv over a 16-wide tile is a 16x16 GEMM with a banded
// Toeplitz B matrix (window 40, K padded to 64 -> 2x v_wmma_f32_16x16x32_bf16
// per (dz,dy) tap plane). B rows K>=40 are exact zeros so the A tail may hold
// arbitrary *finite* padded data.
//
// z-register-blocking: each wave produces TWO adjacent z output tiles; the A
// fragment from padded plane z+zp serves tap dz=zp for slice z and dz=zp-1
// for slice z+1 -> every A load feeds 4 WMMAs.
//
// B-register-carry: with dy outer / zp inner, B(dz=zp-1) needed by c1 is the
// B(dz) c0 used last iteration -> carried in registers, halving B loads.
// Net: 4 A loads + 4 B loads (16B each, L2-resident) per 4 WMMAs.
// ---------------------------------------------------------------------------

typedef __attribute__((ext_vector_type(16))) __bf16 v16bf;
typedef __attribute__((ext_vector_type(8)))  __bf16 v8bf;
typedef __attribute__((ext_vector_type(8)))  float  v8f;

#define KS   25
#define NR   36      // PSF_half radial length
#define NB   2
#define NZ   64
#define NY   256
#define NX   256
#define PADD 88      // 64 + 24
#define PADH 280     // 256 + 24
#define PADW 320     // 256 + 24 rounded to 320; zero tail keeps K=40..63 reads finite
#define PLANE ((long long)PADH * PADW)

// ---------------------------------------------------------------------------
// Kernel 1: build normalized PSF and pre-swizzled Toeplitz B fragments.
// bmat layout: slot = (p*2 + f)*32 + lane, 16 bf16 per slot  (p = dz*25+dy).
// Per-lane WMMA B (32x16 bf16) layout: col n = lane&15,
//   element j = B[K = f*32 + (lane>=16?8:0) + (j<8?j:j+8)][n],  B[K][n] = w[K-n].
// ---------------------------------------------------------------------------
__global__ void build_psf_bmat(const float* __restrict__ psf_half,
                               __bf16* __restrict__ bmat)
{
    __shared__ float psf[KS * KS * KS];   // 62.5 KB of the 320KB/WGP LDS
    __shared__ float red[256];
    const int tid = threadIdx.x;

    for (int i = tid; i < KS * KS * KS; i += 256) {
        int x = i % KS;
        int y = (i / KS) % KS;
        int z = i / (KS * KS);
        float fx = (float)x - 12.0f;
        float fy = (float)y - 12.0f;
        float r2 = sqrtf(fx * fx + fy * fy) * 2.0f;   // rPixel * OS
        int   i1 = (int)floorf(r2);                   // <= 33, i1+1 <= 34 < 36
        float dis = r2 - (float)i1;
        psf[i] = psf_half[z * NR + i1 + 1] * dis +
                 psf_half[z * NR + i1] * (1.0f - dis);
    }
    __syncthreads();

    float s = 0.0f;
    for (int i = tid; i < KS * KS * KS; i += 256) s += psf[i];
    red[tid] = s;
    __syncthreads();
    for (int off = 128; off > 0; off >>= 1) {
        if (tid < off) red[tid] += red[tid + off];
        __syncthreads();
    }
    const float inv = 1.0f / red[0];

    for (int slot = tid; slot < 625 * 2 * 32; slot += 256) {
        int lane = slot & 31;
        int f    = (slot >> 5) & 1;
        int p    = slot >> 6;
        int dz   = p / 25;
        int dy   = p % 25;
        int n     = lane & 15;
        int kbase = f * 32 + ((lane >= 16) ? 8 : 0);
        __bf16* dst = bmat + (long long)slot * 16;
        #pragma unroll
        for (int j = 0; j < 16; ++j) {
            int K   = kbase + (j < 8 ? j : j + 8);
            int dxk = K - n;
            float w = (dxk >= 0 && dxk < KS)
                        ? psf[(dz * KS + dy) * KS + dxk] * inv : 0.0f;
            dst[j] = (__bf16)w;
        }
    }
}

// ---------------------------------------------------------------------------
// Kernel 2: reflect-pad + f32 -> bf16 conversion into [NB][88][280][320].
// ---------------------------------------------------------------------------
__device__ __forceinline__ int reflect_idx(int i, int n)
{
    if (i < 0)       i = -i;            // jnp 'reflect': -1 -> 1
    else if (i >= n) i = 2 * n - 2 - i; //               n  -> n-2
    return i;
}

__global__ void pad_bf16(const float* __restrict__ x, __bf16* __restrict__ xp)
{
    long long idx = (long long)blockIdx.x * blockDim.x + threadIdx.x;
    const long long total = (long long)NB * PADD * PADH * PADW;
    if (idx >= total) return;
    int xpix = (int)(idx % PADW);
    long long t = idx / PADW;
    int ypix = (int)(t % PADH); t /= PADH;
    int zpix = (int)(t % PADD);
    int b    = (int)(t / PADD);
    float v = 0.0f;
    if (xpix < NX + 24) {
        int sx = reflect_idx(xpix - 12, NX);
        int sy = reflect_idx(ypix - 12, NY);
        int sz = reflect_idx(zpix - 12, NZ);
        v = x[(((long long)b * NZ + sz) * NY + sy) * NX + sx];
    }
    xp[idx] = (__bf16)v;
}

// ---------------------------------------------------------------------------
// Kernel 3: conv. One wave -> 16x16 (y,x) tile for TWO adjacent z slices.
// ---------------------------------------------------------------------------
__device__ __forceinline__ v8f wmma_bf16(v16bf a, v16bf b, v8f c)
{
    return __builtin_amdgcn_wmma_f32_16x16x32_bf16(
               false, a, false, b, (short)0, c, false, false);
}

union frag { v16bf v; v8bf h[2]; };

__device__ __forceinline__ void load_a(frag& a1, frag& a2,
                                       const __bf16* __restrict__ rowp)
{
    a1.h[0] = *(const v8bf*)(rowp);          // K base+0..7
    a1.h[1] = *(const v8bf*)(rowp + 16);     // K 16+base..
    a2.h[0] = *(const v8bf*)(rowp + 32);     // K 32+base (>=40 x B=0)
    a2.h[1] = *(const v8bf*)(rowp + 48);     // K 48+base (B=0)
}

__device__ __forceinline__ void load_b(frag& b1, frag& b2,
                                       const __bf16* __restrict__ q)
{
    b1.h[0] = *(const v8bf*)(q);
    b1.h[1] = *(const v8bf*)(q + 8);
    b2.h[0] = *(const v8bf*)(q + 512);       // fragment 1 (K=32..63)
    b2.h[1] = *(const v8bf*)(q + 520);
}

__global__ void conv_wmma(const __bf16* __restrict__ xp,
                          const __bf16* __restrict__ bmat,
                          float* __restrict__ out)
{
    const int lane = threadIdx.x & 31;
    const int wave = threadIdx.x >> 5;          // wave32
    const int tile = blockIdx.x * 8 + wave;     // 16384 dual-z tiles

    const int txi = tile & 15;                  // x tile
    const int tyi = (tile >> 4) & 15;           // y tile
    const int zt  = (tile >> 8) & 31;           // z pair
    const int b   = tile >> 13;
    const int x0  = txi * 16;
    const int y0  = tyi * 16;
    const int z   = zt * 2;

    const int abase = (lane >= 16) ? 8 : 0;
    const int arow  = y0 + (lane & 15);
    const int xoff  = x0 + abase;

    const __bf16* zbase = xp + ((long long)b * PADD + z) * PLANE;
    const __bf16* bp    = bmat + (long long)lane * 16;

    v8f c0 = {};                                // output slice z
    v8f c1 = {};                                // output slice z+1

    #pragma unroll 1
    for (int dy = 0; dy < KS; ++dy) {
        // Row of this (dy) in padded plane 0 of the window.
        const __bf16* rowbase = zbase + (long long)(arow + dy) * PADW + xoff;
        // B(dz=0, dy), fragments for dz advance by 25*1024 elements.
        const __bf16* bq = bp + (long long)dy * 1024;

        frag a1, a2, b1, b2, p1, p2;

        // Pull next dy row (plane 0) toward the WGP.
        __builtin_prefetch(rowbase + PADW, 0, 1);

        // zp = 0: feeds c0 only (dz=0).
        load_b(b1, b2, bq);
        load_a(a1, a2, rowbase);
        c0 = wmma_bf16(a1.v, b1.v, c0);
        c0 = wmma_bf16(a2.v, b2.v, c0);

        // zp = 1..24: c0 uses B(dz=zp); c1 uses carried B(dz=zp-1).
        #pragma unroll 4
        for (int zp = 1; zp < KS; ++zp) {
            p1 = b1;                              // register carry, no reload
            p2 = b2;
            load_b(b1, b2, bq + (long long)zp * (KS * 1024));
            load_a(a1, a2, rowbase + (long long)zp * PLANE);
            c0 = wmma_bf16(a1.v, b1.v, c0);
            c0 = wmma_bf16(a2.v, b2.v, c0);
            c1 = wmma_bf16(a1.v, p1.v, c1);
            c1 = wmma_bf16(a2.v, p2.v, c1);
        }

        // zp = 25: feeds c1 only, with B(dz=24) still in b1,b2.
        load_a(a1, a2, rowbase + (long long)25 * PLANE);
        c1 = wmma_bf16(a1.v, b1.v, c1);
        c1 = wmma_bf16(a2.v, b2.v, c1);
    }

    // C/D layout: N = lane&15 (x), row M = vgpr + (lane>=16 ? 8 : 0) (y)
    const int n     = lane & 15;
    const int mbase = (lane >= 16) ? 8 : 0;
    float* op0 = out + (((long long)b * NZ + z)     * NY + (y0 + mbase)) * NX + x0 + n;
    float* op1 = out + (((long long)b * NZ + z + 1) * NY + (y0 + mbase)) * NX + x0 + n;
    #pragma unroll
    for (int v = 0; v < 8; ++v) {
        op0[(long long)v * NX] = c0[v];
        op1[(long long)v * NX] = c1[v];
    }
}

// ---------------------------------------------------------------------------
// Launch. Workspace: [0, 1.25MB) Bmat bf16 ; [1.25MB.., +31.5MB) padded bf16.
// Total ws required ~ 33 MB.
// ---------------------------------------------------------------------------
extern "C" void kernel_launch(void* const* d_in, const int* in_sizes, int n_in,
                              void* d_out, int out_size, void* d_ws, size_t ws_size,
                              hipStream_t stream)
{
    const float* x        = (const float*)d_in[0];
    const float* psf_half = (const float*)d_in[1];
    float*       out      = (float*)d_out;

    __bf16* bmat = (__bf16*)d_ws;                        // 1,280,000 B
    __bf16* xpad = (__bf16*)((char*)d_ws + 1310720);     // 31,539,200 B

    build_psf_bmat<<<1, 256, 0, stream>>>(psf_half, bmat);

    const long long padTotal = (long long)NB * PADD * PADH * PADW; // 15,769,600
    const int padBlocks = (int)((padTotal + 255) / 256);
    pad_bf16<<<padBlocks, 256, 0, stream>>>(x, xpad);

    conv_wmma<<<2048, 256, 0, stream>>>(xpad, bmat, out);
}
